// DSSM_UBM_60859686584665
// MI455X (gfx1250) — matrix operand order repaired and checked by reading, not compiled, not run on hardware
//
#include <hip/hip_runtime.h>
#include <math.h>

#define B_  128
#define S_  200
#define FL_ 30
#define E_  8
#define NW  2   // waves per block in the attention kernel
#define PPW 4   // (b,s) pairs processed sequentially per wave

typedef __attribute__((ext_vector_type(16))) _Float16 v16h;
typedef __attribute__((ext_vector_type(8)))  float    v8f;
typedef __attribute__((ext_vector_type(4)))  unsigned int u32x4;
typedef __attribute__((ext_vector_type(8)))  int i32x8;
typedef __attribute__((ext_vector_type(4)))  int i32x4;

#if __has_builtin(__builtin_amdgcn_tensor_load_to_lds) && __has_builtin(__builtin_amdgcn_s_wait_tensorcnt)
#define HAVE_TDM 1
#endif

struct KParams {
  // scalar-index inputs (order = setup_inputs order)
  const int *request_wday, *request_hour, *request_min, *uid, *did, *gender, *age, *province;
  const int *vid, *aid, *cate_two, *cate_one, *upload_type, *up_wday, *up_hour, *up_min;
  const int *seq_arr, *seq_mask, *seq_lens, *flow_arr, *flow_mask;
  // embedding tables
  const float *uid_tbl, *did_tbl, *gender_tbl, *age_tbl, *province_tbl;
  const float *vid_tbl, *aid_tbl, *cate2_tbl, *cate1_tbl, *uptype_tbl;
  const float *wday_tbl, *hour_tbl, *min_tbl;
  // weights
  const float *carm_w1, *carm_b1, *carm_w2, *carm_b2;
  const float *ue_w1, *ue_b1, *ue_w2, *ue_b2, *ue_w3, *ue_b3;
  const float *pe_w1, *pe_b1, *pe_w2, *pe_b2, *pe_w3, *pe_b3;
  // workspace + output
  float *ws_rep;        // [B*S][40]
  float *ws_seq;        // [B*S][40]
  float *ws_uin;        // [B][144]
  float *ws_pin;        // [B][64]
  _Float16 *ws_w1h;     // [80][96] f16, transposed + K-padded LDS image of carm_w1
  float *out;           // [B]
};

// ---------------------------------------------------------------------------
// Kernel 0: pre-convert carm_w1 (f32 [k][n]) -> f16 transposed [n][96] image
// ---------------------------------------------------------------------------
__global__ void convert_w1_kernel(const float* __restrict__ w1, _Float16* __restrict__ w1h) {
  int i = blockIdx.x * blockDim.x + threadIdx.x;   // over 80*96
  if (i < 80 * 96) {
    int n = i / 96, k = i % 96;
    w1h[i] = (k < 80) ? (_Float16)w1[k * 80 + n] : (_Float16)0.0f;
  }
}

// ---------------------------------------------------------------------------
// Kernel 1: per-(b,s) DIN attention. One wave32 per (b,s) pair, PPW pairs/wave.
//   din[30,80] = concat(flow_emb, seq_emb) -> relu(din @ W1 + b1) via WMMA
//   logits = h @ w2 + b2, masked softmax over FL, rep = scores @ flow_emb
// ---------------------------------------------------------------------------
__launch_bounds__(NW * 32)
__global__ void carm_attn_kernel(KParams p) {
  __shared__ _Float16 w1t[80][96];          // W1 transposed [n][k], K padded to 96
  __shared__ float    b1s[80], w2s[80];
  __shared__ _Float16 din [NW][32][96];     // A-tile (f16), rows 30..31 zero
  __shared__ float    flow[NW][FL_][40];    // flow embeddings (f32) for weighted sum
  __shared__ float    hbuf[NW][32][80];     // relu(din@W1+b1)
  __shared__ float    seqv[NW][40];         // seq_emb for this (b,s)
  __shared__ float    logitb[NW][32];
  __shared__ float    scoreb[NW][32];

  const int tid  = threadIdx.x;
  const int wv   = tid >> 5;
  const int lane = tid & 31;
  const int hi   = lane >> 4;      // half-wave select
  const int ln   = lane & 15;

  // ---- stage W1 f16 image into LDS: Tensor Data Mover (wave 0 issues) ----
#if defined(HAVE_TDM)
  if (wv == 0) {
    const unsigned lds_addr = (unsigned)(unsigned long long)&w1t[0][0];
    const unsigned long long ga = (unsigned long long)p.ws_w1h;
    u32x4 g0;
    g0.x = 1u;                                          // count=1, user descriptor
    g0.y = lds_addr;                                    // LDS byte address
    g0.z = (unsigned)(ga & 0xffffffffu);                // global_addr[31:0]
    g0.w = (unsigned)((ga >> 32) & 0x01ffffffu)         // global_addr[56:32]
         | (2u << 30);                                  // type=2 ("image")
    i32x8 g1;
    g1[0] = 0x00010000;                 // wg_mask=0, data_size=1 (2 bytes)
    g1[1] = (int)(7680u << 16);         // tensor_dim0[15:0] = 7680 (in [31:16])
    g1[2] = 0x00010000;                 // tensor_dim0 hi = 0, tensor_dim1 = 1
    g1[3] = (int)(7680u << 16);         // tile_dim0 = 7680 (in [31:16])
    g1[4] = 1;                          // tile_dim1 = 1, tile_dim2 = 0
    g1[5] = 7680;                       // tensor_dim0_stride[31:0]
    g1[6] = 0;                          // stride hi / dim1_stride lo
    g1[7] = 0;
    i32x4 z4 = {0, 0, 0, 0};
#if __clang_major__ >= 23
    i32x8 z8 = {0, 0, 0, 0, 0, 0, 0, 0};
    __builtin_amdgcn_tensor_load_to_lds(g0, g1, z4, z4, z8, 0);
#else
    __builtin_amdgcn_tensor_load_to_lds(g0, g1, z4, z4, 0);
#endif
    __builtin_amdgcn_s_wait_tensorcnt(0);
  }
#else
  for (int i = tid; i < (80 * 96 * 2) / 16; i += NW * 32)
    ((uint4*)&w1t[0][0])[i] = ((const uint4*)p.ws_w1h)[i];
#endif
  for (int i = tid; i < 80; i += NW * 32) { b1s[i] = p.carm_b1[i]; w2s[i] = p.carm_w2[i]; }

  const float* tbls[5] = { p.vid_tbl, p.aid_tbl, p.cate2_tbl, p.cate1_tbl, p.uptype_tbl };

  for (int it = 0; it < PPW; ++it) {
    const int pair  = (blockIdx.x * NW + wv) * PPW + it;   // b*S + s
    const long sbase = (long)pair;

    // ---- stage seq_emb[b,s] (40 floats) ----
    for (int j = lane; j < 40; j += 32) {
      int f = j >> 3, e = j & 7;
      int idx = p.seq_arr[sbase * 5 + f];
      seqv[wv][j] = tbls[f][(long)idx * 8 + e];
    }
    __syncthreads();   // also covers W1/b1 staging on first iteration

    // ---- gather flow embeddings + build f16 A-tile (lane m owns row m) ----
    {
      const int m = lane;
      if (m < FL_) {
        long fb = (sbase * FL_ + m) * 5;
        for (int f = 0; f < 5; ++f) {
          int idx = p.flow_arr[fb + f];
          for (int e = 0; e < 8; ++e) {
            float v = tbls[f][(long)idx * 8 + e];
            flow[wv][m][f * 8 + e] = v;
            din [wv][m][f * 8 + e] = (_Float16)v;
          }
        }
        for (int j = 0; j < 40; ++j) din[wv][m][40 + j] = (_Float16)seqv[wv][j];
        for (int j = 80; j < 96; ++j) din[wv][m][j] = (_Float16)0.0f;
      } else {
        for (int j = 0; j < 96; ++j) din[wv][m][j] = (_Float16)0.0f;
      }
    }
    __syncthreads();

    // ---- WMMA GEMM: [32,96] x [96,80] -> acc[2 Mtiles][5 Ntiles] (f32) ----
    v8f acc[2][5] = {};
    for (int kt = 0; kt < 3; ++kt) {
      v16h afrag[2];
      for (int mt = 0; mt < 2; ++mt) {
        // A layout (16-bit 16x32): lane ln row M; halves 0..7 -> K=hi*8+0..7,
        // halves 8..15 -> K=16+hi*8+0..7 (within this K-step)
        const _Float16* r0 = &din[wv][mt * 16 + ln][kt * 32 + hi * 8];
        for (int h = 0; h < 8; ++h) { afrag[mt][h] = r0[h]; afrag[mt][8 + h] = r0[16 + h]; }
      }
      for (int nt = 0; nt < 5; ++nt) {
        // B layout (16-bit 32x16): lane ln col N; halves 0..7 -> K=hi*8+0..7, 8..15 -> K=16+hi*8+0..7
        v16h bfrag;
        const _Float16* c0 = &w1t[nt * 16 + ln][kt * 32 + hi * 8];
        for (int h = 0; h < 8; ++h) { bfrag[h] = c0[h]; bfrag[8 + h] = c0[16 + h]; }
        for (int mt = 0; mt < 2; ++mt)
          acc[mt][nt] = __builtin_amdgcn_wmma_f32_16x16x32_f16(
              false, afrag[mt], false, bfrag, (short)0, acc[mt][nt], false, false);
      }
    }

    // ---- bias + relu, scatter via C/D layout: VGPR r -> M = mt*16 + hi*8 + r ----
    for (int mt = 0; mt < 2; ++mt)
      for (int nt = 0; nt < 5; ++nt)
        for (int r = 0; r < 8; ++r) {
          int m = mt * 16 + hi * 8 + r;
          int n = nt * 16 + ln;
          float v = acc[mt][nt][r] + b1s[n];
          hbuf[wv][m][n] = v > 0.0f ? v : 0.0f;
        }
    __syncthreads();

    // ---- logits (lane m) with mask ----
    {
      float l = -1073741823.0f;  // -2^30 + 1
      if (lane < FL_) {
        float a = p.carm_b2[0];
        for (int n = 0; n < 80; ++n) a += hbuf[wv][lane][n] * w2s[n];
        int mk = p.flow_mask[sbase * FL_ + lane];
        l = (mk != 0) ? a : -1073741823.0f;
      }
      logitb[wv][lane] = l;
    }
    __syncthreads();

    // ---- softmax over FL (all lanes compute same max/sum) ----
    {
      float mx = -3.4e38f;
      for (int m = 0; m < FL_; ++m) mx = fmaxf(mx, logitb[wv][m]);
      float sm = 0.0f;
      for (int m = 0; m < FL_; ++m) sm += __expf(logitb[wv][m] - mx);
      if (lane < FL_) scoreb[wv][lane] = __expf(logitb[wv][lane] - mx) / sm;
    }
    __syncthreads();

    // ---- rep[b,s,e] = sum_m score[m] * flow[m][e]; also emit seq_emb ----
    for (int e = lane; e < 40; e += 32) {
      float r = 0.0f;
      for (int m = 0; m < FL_; ++m) r += scoreb[wv][m] * flow[wv][m][e];
      p.ws_rep[pair * 40 + e] = r;
      p.ws_seq[pair * 40 + e] = seqv[wv][e];
    }
    __syncthreads();   // protect per-wave buffers before next iteration's writes
  }
}

// ---------------------------------------------------------------------------
// Kernel 2: deterministic reduction over S + assemble u_input[144] / p_input[64]
// ---------------------------------------------------------------------------
__global__ void build_inputs_kernel(KParams p) {
  const int b = blockIdx.x, t = threadIdx.x;
  const float len = (float)p.seq_lens[b];
  if (t < 40) {
    float ss = 0.0f, rr = 0.0f;
    for (int s = 0; s < S_; ++s) {
      ss += p.ws_seq[(b * S_ + s) * 40 + t];
      rr += p.ws_rep[(b * S_ + s) * 40 + t];
    }
    p.ws_uin[b * 144 + 64  + t] = ss / len;   // seq_emb_mean
    p.ws_uin[b * 144 + 104 + t] = rr / len;   // rep_mean
  }
  if (t < 64) {
    const int f = t >> 3, e = t & 7;
    const float* utbl[8] = { p.wday_tbl, p.hour_tbl, p.min_tbl, p.uid_tbl,
                             p.did_tbl, p.gender_tbl, p.age_tbl, p.province_tbl };
    const int*   uidx[8] = { p.request_wday, p.request_hour, p.request_min, p.uid,
                             p.did, p.gender, p.age, p.province };
    p.ws_uin[b * 144 + t] = utbl[f][(long)uidx[f][b] * 8 + e];
    const float* ptbl[8] = { p.vid_tbl, p.aid_tbl, p.cate2_tbl, p.cate1_tbl,
                             p.uptype_tbl, p.wday_tbl, p.hour_tbl, p.min_tbl };
    const int*   pidx[8] = { p.vid, p.aid, p.cate_two, p.cate_one,
                             p.upload_type, p.up_wday, p.up_hour, p.up_min };
    p.ws_pin[b * 64 + t] = ptbl[f][(long)pidx[f][b] * 8 + e];
  }
}

// ---------------------------------------------------------------------------
// Kernel 3: both 3-layer MLPs + final dot. One block per batch row.
// ---------------------------------------------------------------------------
__launch_bounds__(128)
__global__ void mlp_dot_kernel(KParams p) {
  __shared__ float uin[144], pin[64];
  __shared__ float uh1[128], ph1[128], uh2[64], ph2[64], uo[32], po[32];
  const int b = blockIdx.x, t = threadIdx.x;
  for (int i = t; i < 144; i += 128) uin[i] = p.ws_uin[b * 144 + i];
  for (int i = t; i < 64;  i += 128) pin[i] = p.ws_pin[b * 64 + i];
  __syncthreads();
  { // layer 1 (128 outputs each)
    float a = p.ue_b1[t];
    for (int k = 0; k < 144; ++k) a += uin[k] * p.ue_w1[k * 128 + t];
    uh1[t] = a > 0.0f ? a : 0.0f;
    float c = p.pe_b1[t];
    for (int k = 0; k < 64; ++k) c += pin[k] * p.pe_w1[k * 128 + t];
    ph1[t] = c > 0.0f ? c : 0.0f;
  }
  __syncthreads();
  if (t < 64) { // layer 2
    float a = p.ue_b2[t];
    for (int k = 0; k < 128; ++k) a += uh1[k] * p.ue_w2[k * 64 + t];
    uh2[t] = a > 0.0f ? a : 0.0f;
    float c = p.pe_b2[t];
    for (int k = 0; k < 128; ++k) c += ph1[k] * p.pe_w2[k * 64 + t];
    ph2[t] = c > 0.0f ? c : 0.0f;
  }
  __syncthreads();
  if (t < 32) { // layer 3 (no relu)
    float a = p.ue_b3[t];
    for (int k = 0; k < 64; ++k) a += uh2[k] * p.ue_w3[k * 32 + t];
    uo[t] = a;
    float c = p.pe_b3[t];
    for (int k = 0; k < 64; ++k) c += ph2[k] * p.pe_w3[k * 32 + t];
    po[t] = c;
  }
  __syncthreads();
  if (t == 0) {
    float d = 0.0f;
    for (int i = 0; i < 32; ++i) d += uo[i] * po[i];
    p.out[b] = d;
  }
}

// ---------------------------------------------------------------------------
extern "C" void kernel_launch(void* const* d_in, const int* in_sizes, int n_in,
                              void* d_out, int out_size, void* d_ws, size_t ws_size,
                              hipStream_t stream) {
  (void)in_sizes; (void)n_in; (void)out_size; (void)ws_size;
  KParams p;
  p.request_wday = (const int*)d_in[0];
  p.request_hour = (const int*)d_in[1];
  p.request_min  = (const int*)d_in[2];
  p.uid          = (const int*)d_in[3];
  p.did          = (const int*)d_in[4];
  p.gender       = (const int*)d_in[5];
  p.age          = (const int*)d_in[6];
  p.province     = (const int*)d_in[7];
  p.vid          = (const int*)d_in[8];
  p.aid          = (const int*)d_in[9];
  p.cate_two     = (const int*)d_in[10];
  p.cate_one     = (const int*)d_in[11];
  p.upload_type  = (const int*)d_in[12];
  p.up_wday      = (const int*)d_in[13];
  p.up_hour      = (const int*)d_in[14];
  p.up_min       = (const int*)d_in[15];
  p.seq_arr      = (const int*)d_in[16];
  p.seq_mask     = (const int*)d_in[17];
  p.seq_lens     = (const int*)d_in[18];
  p.flow_arr     = (const int*)d_in[19];
  p.flow_mask    = (const int*)d_in[20];
  p.uid_tbl      = (const float*)d_in[21];
  p.did_tbl      = (const float*)d_in[22];
  p.gender_tbl   = (const float*)d_in[23];
  p.age_tbl      = (const float*)d_in[24];
  p.province_tbl = (const float*)d_in[25];
  p.vid_tbl      = (const float*)d_in[26];
  p.aid_tbl      = (const float*)d_in[27];
  p.cate2_tbl    = (const float*)d_in[28];
  p.cate1_tbl    = (const float*)d_in[29];
  p.uptype_tbl   = (const float*)d_in[30];
  p.wday_tbl     = (const float*)d_in[31];
  p.hour_tbl     = (const float*)d_in[32];
  p.min_tbl      = (const float*)d_in[33];
  p.carm_w1 = (const float*)d_in[34]; p.carm_b1 = (const float*)d_in[35];
  p.carm_w2 = (const float*)d_in[36]; p.carm_b2 = (const float*)d_in[37];
  p.ue_w1 = (const float*)d_in[38]; p.ue_b1 = (const float*)d_in[39];
  p.ue_w2 = (const float*)d_in[40]; p.ue_b2 = (const float*)d_in[41];
  p.ue_w3 = (const float*)d_in[42]; p.ue_b3 = (const float*)d_in[43];
  p.pe_w1 = (const float*)d_in[44]; p.pe_b1 = (const float*)d_in[45];
  p.pe_w2 = (const float*)d_in[46]; p.pe_b2 = (const float*)d_in[47];
  p.pe_w3 = (const float*)d_in[48]; p.pe_b3 = (const float*)d_in[49];

  float* ws = (float*)d_ws;
  const size_t BS40 = (size_t)B_ * S_ * 40;
  p.ws_rep = ws;
  p.ws_seq = ws + BS40;
  p.ws_uin = ws + 2 * BS40;
  p.ws_pin = ws + 2 * BS40 + (size_t)B_ * 144;
  p.ws_w1h = (_Float16*)(ws + 2 * BS40 + (size_t)B_ * 144 + (size_t)B_ * 64);
  p.out    = (float*)d_out;

  convert_w1_kernel <<<(80 * 96 + 255) / 256, 256, 0, stream>>>(p.carm_w1, p.ws_w1h);
  carm_attn_kernel  <<<(B_ * S_) / (NW * PPW), NW * 32, 0, stream>>>(p);
  build_inputs_kernel<<<B_, 64, 0, stream>>>(p);
  mlp_dot_kernel    <<<B_, 128, 0, stream>>>(p);
}
